// EditDecoderRNN_283467842856
// MI455X (gfx1250) — matrix operand order repaired and checked by reading, not compiled, not run on hardware
//
#include <hip/hip_runtime.h>
#include <stdint.h>
#include <stddef.h>

// ---------------- problem dims ----------------
#define B_   16
#define N_   128
#define S_   160
#define LS_  160
#define T_   127     // N-1
#define V_   21128
#define E_   768
#define H_   768
#define NA_  200
#define FH_  3840    // 5*H

typedef __bf16 bf16_t;
typedef __attribute__((ext_vector_type(16))) __bf16 v16bf;
typedef __attribute__((ext_vector_type(8)))  float  v8f;

// ---------------- small helpers ----------------
__device__ __forceinline__ bf16_t f2bf(float f){
  union { float f; uint32_t u; } v; v.f = f;
  uint32_t r = (v.u + 0x7FFFu + ((v.u >> 16) & 1u)) >> 16;   // RNE
  uint16_t s = (uint16_t)r;
  return __builtin_bit_cast(__bf16, s);
}
__device__ __forceinline__ float bf2f(bf16_t b){
  uint16_t s = __builtin_bit_cast(uint16_t, b);
  union { uint32_t u; float f; } v; v.u = ((uint32_t)s) << 16;
  return v.f;
}
__device__ __forceinline__ float sigmf(float x){ return 1.0f/(1.0f+expf(-x)); }

__device__ __forceinline__ v8f v8f_zero(){
  v8f z;
#pragma unroll
  for (int i=0;i<8;++i) z[i]=0.f;
  return z;
}

// D = A(16x32 bf16) * B(32x16 bf16) + C(16x16 f32)
__device__ __forceinline__ v8f wmma_bf16(v16bf a, v16bf b, v8f c){
  return __builtin_amdgcn_wmma_f32_16x16x32_bf16(false, a, false, b, (short)0, c, false, false);
}

// A fragment: lane m = lane&15, g = lane>>4. Per ISA 16-bit A layout:
// elements 0..7  = A[m, kb + g*8 + j], elements 8..15 = A[m, kb + 16 + g*8 + j]
__device__ __forceinline__ v16bf load_a_row(const bf16_t* rowp, int kb, int g){
  const bf16_t* p = rowp + kb + g*8;
  v16bf r;
#pragma unroll
  for (int j=0;j<8;++j){ r[j]=p[j]; r[j+8]=p[16+j]; }
  return r;
}
// B fragment from weight stored [N,K] row-major: lane n = lane&15, g = lane>>4,
// elements 0..7 = K kb + g*8 + j, elements 8..15 = K kb + 16 + g*8 + j of row (n0+n).
__device__ __forceinline__ v16bf load_b_row(const bf16_t* W, int ldW, int row, int kb, int g){
  const bf16_t* p = W + (size_t)row*ldW + kb + g*8;
  v16bf r;
#pragma unroll
  for (int j=0;j<8;++j){ r[j]=p[j]; r[j+8]=p[16+j]; }
  return r;
}

// ---------------- cast / transpose ----------------
__global__ void __launch_bounds__(256) cast_f32_bf16_kernel(const float* __restrict__ src,
                                                            bf16_t* __restrict__ dst, size_t n){
  for (size_t i = blockIdx.x*256ULL + threadIdx.x; i < n; i += (size_t)gridDim.x*256ULL)
    dst[i] = f2bf(src[i]);
}

__global__ void __launch_bounds__(256) transpose_enc_kernel(const bf16_t* __restrict__ encB,
                                                            bf16_t* __restrict__ encT){
  const size_t total=(size_t)B_*S_*H_;
  for (size_t i=blockIdx.x*256ULL+threadIdx.x; i<total; i+=(size_t)gridDim.x*256ULL){
    size_t b = i/((size_t)S_*H_);
    size_t rem = i - b*(size_t)S_*H_;
    int s = (int)(rem / H_);
    int h = (int)(rem - (size_t)s*H_);
    encT[b*(size_t)H_*S_ + (size_t)h*S_ + s] = encB[i];
  }
}

// ---------------- generic WMMA GEMM ----------------
// C[M,N] = act( A[M,K](bf16,ldA) @ W[N,K]^T (bf16,ldW) + bias[n] + mask[n] )
// block = 256 threads (8 waves); block tile = 32 rows x 128 cols.
// Each wave holds one N-tile with TWO accumulators (rows 0..15 / 16..31 of the
// block tile) so each streamed B-fragment feeds two v_wmma ops (halves W traffic).
// The 32x128 A tile is staged into LDS with GLOBAL_LOAD_ASYNC_TO_LDS_B128
// (CDNA5 async copy, ASYNCcnt) -- per-lane 16B chunks, then s_wait_asynccnt.
__global__ void __launch_bounds__(256) gemm_bf16_kernel(
    const bf16_t* __restrict__ A, int ldA,
    const bf16_t* __restrict__ W, int ldW, int Wrows,
    const float* __restrict__ bias, const float* __restrict__ mask,
    float* __restrict__ Cf, bf16_t* __restrict__ Cb, int ldC,
    int M, int N, int K, int act)
{
  __shared__ bf16_t As[32][136];
  const int tid = threadIdx.x, lane = tid & 31, wave = tid >> 5;
  const int g = lane >> 4, nl = lane & 15;
  const int m0 = blockIdx.y * 32;
  const int n0 = (blockIdx.x * 8 + wave) * 16;
  const int wrow = (n0 + nl) < Wrows ? (n0 + nl) : (Wrows - 1);

  v8f acc0 = v8f_zero();
  v8f acc1 = v8f_zero();

  for (int kc = 0; kc < K; kc += 128){
    // ---- async stage: 32 rows x 256B = 512 x 16B chunks, 2 per thread ----
#pragma unroll
    for (int q = 0; q < 2; ++q){
      int cid = tid*2 + q;
      int row = cid >> 4;          // 0..31
      int ch  = cid & 15;          // 0..15 (16B chunks)
      int ar  = m0 + row; if (ar > M-1) ar = M-1;
      int k   = kc + ch*8; if (k + 8 > K) k = K - 8;   // K multiple of 32 >= 8
      const bf16_t* src = A + (size_t)ar*ldA + k;
      uint32_t ldsAddr = (uint32_t)(uintptr_t)&As[row][ch*8];  // ADDR[31:0] = LDS offset
      asm volatile("global_load_async_to_lds_b128 %0, %1, off"
                   :: "v"(ldsAddr), "v"((uint64_t)(uintptr_t)src)
                   : "memory");
    }
    asm volatile("s_wait_asynccnt 0x0" ::: "memory");
    __syncthreads();

    if (kc + 128 < K)
      __builtin_prefetch(W + (size_t)wrow*ldW + kc + 128, 0, 1);   // global_prefetch

    int krem = K - kc; if (krem > 128) krem = 128;
    for (int kf = 0; kf < krem; kf += 32){
      v16bf a0, a1;
      {
        const bf16_t* p0 = &As[nl][kf + g*8];
        const bf16_t* p1 = &As[16 + nl][kf + g*8];
#pragma unroll
        for (int j = 0; j < 8; ++j){
          a0[j] = p0[j]; a0[j+8] = p0[16+j];
          a1[j] = p1[j]; a1[j+8] = p1[16+j];
        }
      }
      v16bf bb = load_b_row(W, ldW, wrow, kc + kf, g);
      acc0 = wmma_bf16(a0, bb, acc0);
      acc1 = wmma_bf16(a1, bb, acc1);
    }
    __syncthreads();
  }

  const int n = n0 + nl;
#pragma unroll
  for (int r = 0; r < 8; ++r){
    int m = m0 + r + g*8;
    if (m < M && n < N){
      float v = acc0[r];
      if (bias) v += bias[n];
      if (mask) v += mask[n];
      if (act == 1) v = tanhf(v);
      if (Cf) Cf[(size_t)m*ldC + n] = v;
      if (Cb) Cb[(size_t)m*ldC + n] = f2bf(v);
    }
    int m2 = m + 16;
    if (m2 < M && n < N){
      float v = acc1[r];
      if (bias) v += bias[n];
      if (mask) v += mask[n];
      if (act == 1) v = tanhf(v);
      if (Cf) Cf[(size_t)m2*ldC + n] = v;
      if (Cb) Cb[(size_t)m2*ldC + n] = f2bf(v);
    }
  }
}

// ---------------- LSTM (persistent workgroup per LSTM) ----------------
// block 512 = 16 waves; wave handles 3 j-tiles (48 tiles of H=768);
// z = emb[tok]@Wih^T + h@Whh^T + b  computed with WMMA; h kept in LDS (bf16),
// c kept in registers in C-fragment layout.
__global__ void __launch_bounds__(512) lstm_kernel(
    const int* __restrict__ tokE, const int* __restrict__ tokA, const int* __restrict__ tokW,
    const bf16_t* __restrict__ emb,
    const bf16_t* WihE, const bf16_t* WhhE, const float* bihE, const float* bhhE,
    const bf16_t* WihA, const bf16_t* WhhA, const float* bihA, const float* bhhA,
    const bf16_t* WihW, const bf16_t* WhhW, const float* bihW, const float* bhhW,
    const float* __restrict__ h0, const float* __restrict__ c0,
    bf16_t* __restrict__ outE, bf16_t* __restrict__ outA, bf16_t* __restrict__ outW)
{
  const int lid = blockIdx.x;
  const int* tok; const bf16_t* Wih; const bf16_t* Whh; const float* bih; const float* bhh;
  bf16_t* outp; int steps;
  if (lid == 0){ tok=tokE; Wih=WihE; Whh=WhhE; bih=bihE; bhh=bhhE; outp=outE; steps=N_;  }
  else if (lid == 1){ tok=tokA; Wih=WihA; Whh=WhhA; bih=bihA; bhh=bhhA; outp=outA; steps=N_;  }
  else { tok=tokW; Wih=WihW; Whh=WhhW; bih=bihW; bhh=bhhW; outp=outW; steps=LS_; }

  __shared__ bf16_t hs[16][H_ + 8];
  __shared__ float  bsum[4*H_];

  const int tid = threadIdx.x, lane = tid & 31, wave = tid >> 5;
  const int g = lane >> 4, nl = lane & 15;

  for (int i = tid; i < 4*H_; i += 512) bsum[i] = bih[i] + bhh[i];
  for (int i = tid; i < 16*H_; i += 512){ int m = i / H_, j = i - m*H_; hs[m][j] = f2bf(h0[m*H_ + j]); }

  float c[3][8];
#pragma unroll
  for (int i = 0; i < 3; ++i){
    int j = (wave*3 + i)*16 + nl;
#pragma unroll
    for (int r = 0; r < 8; ++r){ int m = r + g*8; c[i][r] = c0[m*H_ + j]; }
  }
  __syncthreads();

  for (int t = 0; t < steps; ++t){
    // fused embedding gather: each lane's A row is emb[tok[m,t]]
    const bf16_t* arow = emb + (size_t)tok[nl*steps + t]*E_;

    v8f acc[3][4];
#pragma unroll
    for (int i = 0; i < 3; ++i)
#pragma unroll
      for (int q = 0; q < 4; ++q){
        float bv = bsum[q*H_ + (wave*3 + i)*16 + nl];
#pragma unroll
        for (int r = 0; r < 8; ++r) acc[i][q][r] = bv;
      }

    // x @ Wih^T
    for (int kt = 0; kt < E_/32; ++kt){
      const int kb = kt*32;
      v16bf a = load_a_row(arow, kb, g);
#pragma unroll
      for (int i = 0; i < 3; ++i)
#pragma unroll
        for (int q = 0; q < 4; ++q){
          v16bf bb = load_b_row(Wih, E_, q*H_ + (wave*3 + i)*16 + nl, kb, g);
          acc[i][q] = wmma_bf16(a, bb, acc[i][q]);
        }
    }
    // h @ Whh^T (h from LDS)
    for (int kt = 0; kt < H_/32; ++kt){
      const int kb = kt*32;
      v16bf a;
      {
        const bf16_t* p = &hs[nl][kb + g*8];
#pragma unroll
        for (int j = 0; j < 8; ++j){ a[j] = p[j]; a[j+8] = p[16+j]; }
      }
#pragma unroll
      for (int i = 0; i < 3; ++i)
#pragma unroll
        for (int q = 0; q < 4; ++q){
          v16bf bb = load_b_row(Whh, H_, q*H_ + (wave*3 + i)*16 + nl, kb, g);
          acc[i][q] = wmma_bf16(a, bb, acc[i][q]);
        }
    }

    float hnew[3][8];
#pragma unroll
    for (int i = 0; i < 3; ++i)
#pragma unroll
      for (int r = 0; r < 8; ++r){
        float iv = sigmf(acc[i][0][r]);
        float fv = sigmf(acc[i][1][r]);
        float gv = tanhf(acc[i][2][r]);
        float ov = sigmf(acc[i][3][r]);
        c[i][r] = fv*c[i][r] + iv*gv;
        hnew[i][r] = ov*tanhf(c[i][r]);
      }

    __syncthreads();   // all waves done reading old h
#pragma unroll
    for (int i = 0; i < 3; ++i){
      int j = (wave*3 + i)*16 + nl;
#pragma unroll
      for (int r = 0; r < 8; ++r){
        int m = r + g*8;
        bf16_t hb = f2bf(hnew[i][r]);
        hs[m][j] = hb;
        outp[((size_t)m*steps + t)*H_ + j] = hb;
      }
    }
    __syncthreads();   // new h visible before next step
  }
}

// ---------------- counters (scalar scans, lane-per-batch) ----------------
__global__ void counters_kernel(const int* __restrict__ edits, const int* __restrict__ org,
                                int* __restrict__ kd, int* __restrict__ ki, int* __restrict__ an){
  int b = threadIdx.x;
  if (b >= B_) return;
  int kdacc = 0, kiacc = 0, a = 0;
  for (int t = 0; t < T_; ++t){
    int gold = edits[b*N_ + t + 1];
    kd[b*T_ + t] = kdacc;
    ki[b*T_ + t] = kiacc;
    an[b*T_ + t] = a;
    int inc_kd = (gold == 1 || gold == 2) ? 1 : 0;
    int inc_ki = (gold != 2 && gold != 102 && gold != 0) ? 1 : 0;
    kdacc += inc_kd;
    kiacc += inc_ki;
    int nxt = a + 1;
    int nc  = nxt < S_ ? nxt : (S_ - 1);
    int tokv = org[b*S_ + nc];
    bool cond = (gold != 2) && (gold != 102) && (gold != 0) && (gold != 1) &&
                (nxt < S_) && (tokv == 103 || tokv == 3 || tokv == 4);
    a = cond ? nxt : (kdacc > a ? kdacc : a);
  }
}

// ---------------- contexts + ref-softmax; builds feature matrices ----------------
__global__ void __launch_bounds__(256) context_kernel(
    const bf16_t* __restrict__ outE, const bf16_t* __restrict__ outA,
    const bf16_t* __restrict__ outW, const bf16_t* __restrict__ encB,
    const float* __restrict__ Wref, const float* __restrict__ bref,
    const int* __restrict__ kd, const int* __restrict__ ki, const int* __restrict__ an,
    bf16_t* __restrict__ featE, bf16_t* __restrict__ featA)
{
  const int bt = blockIdx.x;
  const int b = bt / T_, t = bt - b*T_;
  const int tid = threadIdx.x;
  const bf16_t* de = outE + ((size_t)b*N_ + t)*H_;
  const bf16_t* da = outA + ((size_t)b*N_ + t)*H_;
  const int kdv = kd[bt], kiv = ki[bt], anv = an[bt];
  const bf16_t* ci = encB + ((size_t)b*S_ + kdv)*H_;
  const bf16_t* ca = encB + ((size_t)b*S_ + anv)*H_;
  const bf16_t* cw = outW + ((size_t)b*LS_ + kiv)*H_;

  float p[4] = {0.f,0.f,0.f,0.f};
  for (int h = tid; h < H_; h += 256){
    float w0 = Wref[h], w1 = Wref[H_ + h];
    float deh = bf2f(de[h]), dah = bf2f(da[h]);
    p[0] += deh*w0; p[1] += deh*w1; p[2] += dah*w0; p[3] += dah*w1;
  }
  __shared__ float red[256];
  __shared__ float dots[4];
  for (int d = 0; d < 4; ++d){
    red[tid] = p[d]; __syncthreads();
    for (int s2 = 128; s2 > 0; s2 >>= 1){ if (tid < s2) red[tid] += red[tid + s2]; __syncthreads(); }
    if (tid == 0) dots[d] = red[0];
    __syncthreads();
  }
  float e0 = dots[0] + bref[0], e1 = dots[1] + bref[1];
  float m  = fmaxf(e0, e1); float x0 = expf(e0 - m), x1 = expf(e1 - m); float inv = 1.f/(x0 + x1);
  const float we0 = x0*inv, we1 = x1*inv;
  e0 = dots[2] + bref[0]; e1 = dots[3] + bref[1];
  m = fmaxf(e0, e1); x0 = expf(e0 - m); x1 = expf(e1 - m); inv = 1.f/(x0 + x1);
  const float wa0 = x0*inv, wa1 = x1*inv;

  bf16_t* fe = featE + (size_t)bt*FH_;
  bf16_t* fa = featA + (size_t)bt*FH_;
  for (int h = tid; h < H_; h += 256){
    bf16_t deb = de[h], dab = da[h], cwb = cw[h];
    float cif = bf2f(ci[h]), caf = bf2f(ca[h]);
    fe[h]           = deb;
    fe[H_   + h]    = dab;
    fe[3*H_ + h]    = f2bf(we0*cif + we1*caf);   // c_edit
    fe[4*H_ + h]    = cwb;
    fa[h]           = dab;
    fa[H_   + h]    = deb;
    fa[3*H_ + h]    = f2bf(wa0*cif + wa1*caf);   // c_action
    fa[4*H_ + h]    = cwb;
    // [2H:3H) slice (applied) filled by applied_kernel
  }
}

// ---------------- batched attention logits: key[b] @ enc[b]^T ----------------
__global__ void __launch_bounds__(256) attn_logits_kernel(
    const bf16_t* __restrict__ keyB, const bf16_t* __restrict__ encB, float* __restrict__ attnF)
{
  const int b = blockIdx.x, mt = blockIdx.y;
  const int tid = threadIdx.x, lane = tid & 31, wave = tid >> 5;
  const int g = lane >> 4, nl = lane & 15;
  int trow = mt*16 + nl; if (trow > T_-1) trow = T_-1;
  const bf16_t* arow = keyB + ((size_t)b*T_ + trow)*H_;
  const bf16_t* Wb   = encB + (size_t)b*S_*H_;        // rows = s, [N,K] layout
  for (int nt = wave; nt < S_/16; nt += 8){
    v8f acc = v8f_zero();
    for (int kt = 0; kt < H_/32; ++kt){
      v16bf a  = load_a_row(arow, kt*32, g);
      v16bf bb = load_b_row(Wb, H_, nt*16 + nl, kt*32, g);
      acc = wmma_bf16(a, bb, acc);
    }
#pragma unroll
    for (int r = 0; r < 8; ++r){
      int t = mt*16 + r + g*8;
      if (t < T_) attnF[((size_t)b*T_ + t)*S_ + nt*16 + nl] = acc[r];
    }
  }
}

__global__ void __launch_bounds__(256) attn_softmax_kernel(const float* __restrict__ attnF,
                                                           bf16_t* __restrict__ attnB){
  const int r = blockIdx.x, tid = threadIdx.x;
  const float* x = attnF + (size_t)r*S_;
  __shared__ float red[256];
  float v = (tid < S_) ? x[tid] : -3.0e38f;
  red[tid] = v; __syncthreads();
  for (int s2 = 128; s2 > 0; s2 >>= 1){ if (tid < s2) red[tid] = fmaxf(red[tid], red[tid+s2]); __syncthreads(); }
  float mx = red[0]; __syncthreads();
  float e = (tid < S_) ? expf(x[tid] - mx) : 0.f;
  red[tid] = e; __syncthreads();
  for (int s2 = 128; s2 > 0; s2 >>= 1){ if (tid < s2) red[tid] += red[tid+s2]; __syncthreads(); }
  float inv = 1.f/red[0];
  if (tid < S_) attnB[(size_t)r*S_ + tid] = f2bf(e*inv);
}

// ---------------- applied = attn[b] @ enc[b], writes feat slice [2H:3H) ----------------
__global__ void __launch_bounds__(256) applied_kernel(
    const bf16_t* __restrict__ attnB, const bf16_t* __restrict__ encT,
    bf16_t* __restrict__ featE, bf16_t* __restrict__ featA)
{
  const int b = blockIdx.x, mt = blockIdx.y;
  const int tid = threadIdx.x, lane = tid & 31, wave = tid >> 5;
  const int g = lane >> 4, nl = lane & 15;
  int trow = mt*16 + nl; if (trow > T_-1) trow = T_-1;
  const bf16_t* arow = attnB + ((size_t)b*T_ + trow)*S_;
  const bf16_t* Wb   = encT  + (size_t)b*H_*S_;       // rows = h, ld = S
  for (int nt = wave; nt < H_/16; nt += 8){
    v8f acc = v8f_zero();
    for (int kt = 0; kt < S_/32; ++kt){
      v16bf a  = load_a_row(arow, kt*32, g);
      v16bf bb = load_b_row(Wb, S_, nt*16 + nl, kt*32, g);
      acc = wmma_bf16(a, bb, acc);
    }
#pragma unroll
    for (int r = 0; r < 8; ++r){
      int t = mt*16 + r + g*8;
      if (t < T_){
        size_t ro = ((size_t)b*T_ + t)*FH_ + 2*H_ + nt*16 + nl;
        bf16_t v = f2bf(acc[r]);
        featE[ro] = v; featA[ro] = v;
      }
    }
  }
}

// ---------------- row log-softmax (in place on fp32) ----------------
__global__ void __launch_bounds__(256) logsoftmax_kernel(float* __restrict__ out, int len){
  const int r = blockIdx.x, tid = threadIdx.x;
  float* row = out + (size_t)r*len;
  __shared__ float red[256];
  float mx = -3.0e38f;
  for (int i = tid; i < len; i += 256) mx = fmaxf(mx, row[i]);
  red[tid] = mx; __syncthreads();
  for (int s2 = 128; s2 > 0; s2 >>= 1){ if (tid < s2) red[tid] = fmaxf(red[tid], red[tid+s2]); __syncthreads(); }
  mx = red[0]; __syncthreads();
  float s = 0.f;
  for (int i = tid; i < len; i += 256) s += expf(row[i] - mx);
  red[tid] = s; __syncthreads();
  for (int s2 = 128; s2 > 0; s2 >>= 1){ if (tid < s2) red[tid] += red[tid+s2]; __syncthreads(); }
  float lse = mx + logf(red[0]);
  __syncthreads();
  for (int i = tid; i < len; i += 256) row[i] -= lse;
}

// ---------------- host: orchestrate ----------------
extern "C" void kernel_launch(void* const* d_in, const int* in_sizes, int n_in,
                              void* d_out, int out_size, void* d_ws, size_t ws_size,
                              hipStream_t stream) {
  (void)in_sizes; (void)n_in; (void)out_size; (void)ws_size;

  const int*   input_edits   = (const int*)  d_in[0];
  const int*   input_actions = (const int*)  d_in[1];
  const float* h0            = (const float*)d_in[2];
  const float* c0            = (const float*)d_in[3];
  const float* enc_org       = (const float*)d_in[4];
  const int*   org_ids       = (const int*)  d_in[5];
  const int*   simp_sent     = (const int*)  d_in[6];
  // params (dict insertion order)
  const float* emb      = (const float*)d_in[7];
  const float* W_align  = (const float*)d_in[8];
  const float* Wp       = (const float*)d_in[9];
  const float* Wref     = (const float*)d_in[10];
  const float* bref     = (const float*)d_in[11];
  const float* Wmlp     = (const float*)d_in[12];
  const float* bmlp     = (const float*)d_in[13];
  const float* Wact     = (const float*)d_in[14];
  const float* bact     = (const float*)d_in[15];
  /* Wout == emb (d_in[16]), bout d_in[17], Wouta == emb[:NA] (d_in[18]), bouta d_in[19] */
  const float* bout     = (const float*)d_in[17];
  const float* bouta    = (const float*)d_in[19];
  const float* Wih_e    = (const float*)d_in[20];
  const float* Whh_e    = (const float*)d_in[21];
  const float* bih_e    = (const float*)d_in[22];
  const float* bhh_e    = (const float*)d_in[23];
  const float* Wih_a    = (const float*)d_in[24];
  const float* Whh_a    = (const float*)d_in[25];
  const float* bih_a    = (const float*)d_in[26];
  const float* bhh_a    = (const float*)d_in[27];
  const float* Wih_w    = (const float*)d_in[28];
  const float* Whh_w    = (const float*)d_in[29];
  const float* bih_w    = (const float*)d_in[30];
  const float* bhh_w    = (const float*)d_in[31];
  const float* action_mask = (const float*)d_in[32];

  // ---- workspace layout ----
  uint8_t* ws = (uint8_t*)d_ws;
  size_t off = 0;
  auto take = [&](size_t bytes) -> void* {
    void* p = ws + off;
    off += (bytes + 255) & ~(size_t)255;
    return p;
  };
  bf16_t* embB    = (bf16_t*)take((size_t)V_*E_*2);
  bf16_t* WalignB = (bf16_t*)take((size_t)H_*E_*2);
  bf16_t* WpB     = (bf16_t*)take((size_t)H_*H_*2);
  bf16_t* WmlpB   = (bf16_t*)take((size_t)E_*FH_*2);
  bf16_t* WactB   = (bf16_t*)take((size_t)E_*FH_*2);
  bf16_t* WihEB   = (bf16_t*)take((size_t)4*H_*E_*2);
  bf16_t* WhhEB   = (bf16_t*)take((size_t)4*H_*H_*2);
  bf16_t* WihAB   = (bf16_t*)take((size_t)4*H_*E_*2);
  bf16_t* WhhAB   = (bf16_t*)take((size_t)4*H_*H_*2);
  bf16_t* WihWB   = (bf16_t*)take((size_t)4*H_*E_*2);
  bf16_t* WhhWB   = (bf16_t*)take((size_t)4*H_*H_*2);
  bf16_t* encInB  = (bf16_t*)take((size_t)B_*S_*E_*2);
  bf16_t* encB    = (bf16_t*)take((size_t)B_*S_*H_*2);
  bf16_t* encTB   = (bf16_t*)take((size_t)B_*H_*S_*2);
  bf16_t* outEB   = (bf16_t*)take((size_t)B_*N_*H_*2);
  bf16_t* outAB   = (bf16_t*)take((size_t)B_*N_*H_*2);
  bf16_t* outWB   = (bf16_t*)take((size_t)B_*LS_*H_*2);
  int*    kdBuf   = (int*)   take((size_t)B_*T_*4);
  int*    kiBuf   = (int*)   take((size_t)B_*T_*4);
  int*    anBuf   = (int*)   take((size_t)B_*T_*4);
  bf16_t* featE   = (bf16_t*)take((size_t)B_*T_*FH_*2);
  bf16_t* featA   = (bf16_t*)take((size_t)B_*T_*FH_*2);
  bf16_t* keyB    = (bf16_t*)take((size_t)B_*T_*H_*2);
  float*  attnF   = (float*) take((size_t)B_*T_*S_*4);
  bf16_t* attnB   = (bf16_t*)take((size_t)B_*T_*S_*2);
  bf16_t* oaB     = (bf16_t*)take((size_t)B_*T_*H_*2);
  bf16_t* oeB     = (bf16_t*)take((size_t)B_*T_*H_*2);

  auto castN = [&](const float* s, bf16_t* d, size_t n){
    unsigned blocks = (unsigned)((n + 255) / 256);
    if (blocks > 4096u) blocks = 4096u;
    cast_f32_bf16_kernel<<<blocks, 256, 0, stream>>>(s, d, n);
  };

  // ---- 1. weight/input casts ----
  castN(emb,     embB,    (size_t)V_*E_);
  castN(W_align, WalignB, (size_t)H_*E_);
  castN(Wp,      WpB,     (size_t)H_*H_);
  castN(Wmlp,    WmlpB,   (size_t)E_*FH_);
  castN(Wact,    WactB,   (size_t)E_*FH_);
  castN(Wih_e,   WihEB,   (size_t)4*H_*E_);
  castN(Whh_e,   WhhEB,   (size_t)4*H_*H_);
  castN(Wih_a,   WihAB,   (size_t)4*H_*E_);
  castN(Whh_a,   WhhAB,   (size_t)4*H_*H_);
  castN(Wih_w,   WihWB,   (size_t)4*H_*E_);
  castN(Whh_w,   WhhWB,   (size_t)4*H_*H_);
  castN(enc_org, encInB,  (size_t)B_*S_*E_);

  // ---- 2. enc = tanh(X @ W_align^T) : M=2560, N=768, K=768 ----
  gemm_bf16_kernel<<<dim3(6, (B_*S_)/32), 256, 0, stream>>>(
      encInB, E_, WalignB, E_, H_, nullptr, nullptr,
      nullptr, encB, H_, B_*S_, H_, E_, /*act=*/1);
  transpose_enc_kernel<<<4096, 256, 0, stream>>>(encB, encTB);

  // ---- 3. three LSTMs (one workgroup each) ----
  lstm_kernel<<<3, 512, 0, stream>>>(
      input_edits, input_actions, simp_sent, embB,
      WihEB, WhhEB, bih_e, bhh_e,
      WihAB, WhhAB, bih_a, bhh_a,
      WihWB, WhhWB, bih_w, bhh_w,
      h0, c0, outEB, outAB, outWB);

  // ---- 4. counters ----
  counters_kernel<<<1, 32, 0, stream>>>(input_edits, org_ids, kdBuf, kiBuf, anBuf);

  // ---- 5. contexts + ref softmax -> feature matrices ----
  context_kernel<<<B_*T_, 256, 0, stream>>>(
      outEB, outAB, outWB, encB, Wref, bref, kdBuf, kiBuf, anBuf, featE, featA);

  const int mt32 = (B_*T_ + 31) / 32;   // 64 row-tiles of 32 for M=2032

  // ---- 6. key = c_word @ Wp^T (A = featE[:, 4H:5H]) ----
  gemm_bf16_kernel<<<dim3(6, mt32), 256, 0, stream>>>(
      featE + 4*H_, FH_, WpB, H_, H_, nullptr, nullptr,
      nullptr, keyB, H_, B_*T_, H_, H_, 0);

  // ---- 7. attention ----
  attn_logits_kernel<<<dim3(B_, 8), 256, 0, stream>>>(keyB, encB, attnF);
  attn_softmax_kernel<<<B_*T_, 256, 0, stream>>>(attnF, attnB);
  applied_kernel<<<dim3(B_, 8), 256, 0, stream>>>(attnB, encTB, featE, featA);

  // ---- 8. feature MLPs (K = 5H = 3840) ----
  gemm_bf16_kernel<<<dim3(6, mt32), 256, 0, stream>>>(
      featA, FH_, WactB, FH_, E_, bact, nullptr,
      nullptr, oaB, H_, B_*T_, E_, FH_, 1);
  gemm_bf16_kernel<<<dim3(6, mt32), 256, 0, stream>>>(
      featE, FH_, WmlpB, FH_, E_, bmlp, nullptr,
      nullptr, oeB, H_, B_*T_, E_, FH_, 1);

  // ---- 9. output projections into d_out ----
  float* editOut = (float*)d_out;                       // (B*T, V)
  float* actOut  = editOut + (size_t)B_*T_*V_;          // (B*T, NA)
  // action logits: Wouta = emb[:NA] -> reuse embB rows
  gemm_bf16_kernel<<<dim3(2, mt32), 256, 0, stream>>>(
      oaB, H_, embB, E_, NA_, bouta, action_mask,
      actOut, nullptr, NA_, B_*T_, NA_, H_, 0);
  // edit logits: Wout = emb -> embB, N = V (1321 n-tiles -> grid.x = 166)
  gemm_bf16_kernel<<<dim3(166, mt32), 256, 0, stream>>>(
      oeB, H_, embB, E_, V_, bout, nullptr,
      editOut, nullptr, V_, B_*T_, V_, H_, 0);

  // ---- 10. log-softmax in place ----
  logsoftmax_kernel<<<B_*T_, 256, 0, stream>>>(actOut, NA_);
  logsoftmax_kernel<<<B_*T_, 256, 0, stream>>>(editOut, V_);
}
